// VADGuidedCrossAttention_31696858644813
// MI455X (gfx1250) — matrix operand order repaired and checked by reading, not compiled, not run on hardware
//
#include <hip/hip_runtime.h>
#include <math.h>

// ---------------------------------------------------------------------------
// Types for CDNA5 WMMA (wave32, 16x16x32 bf16 -> f32)
// ---------------------------------------------------------------------------
typedef __bf16 bf16;
typedef __attribute__((ext_vector_type(8)))  bf16  bf16x8;
typedef __attribute__((ext_vector_type(16))) bf16  bf16x16;
typedef __attribute__((ext_vector_type(8)))  float f32x8;

// Feature gate: the async-load builtin existing proves gfx1250 has the
// GLOBAL_LOAD_ASYNC_TO_LDS path; we emit the instruction via inline asm to
// sidestep the builtin's CUDA-address-space-qualified pointer parameters.
#if defined(__gfx1250__) && __has_builtin(__builtin_amdgcn_global_load_async_to_lds_b128)
#define USE_ASYNC_LDS 1
#else
#define USE_ASYNC_LDS 0
#endif

__device__ __forceinline__ f32x8 wmma_bf16(bf16x16 a, bf16x16 b, f32x8 c) {
  return __builtin_amdgcn_wmma_f32_16x16x32_bf16(
      /*neg_a=*/false, a, /*neg_b=*/false, b,
      /*c_mod=*/(short)0, c, /*reuse_a=*/false, /*reuse_b=*/false);
}

// A fragment (16x32, MxK): lane r (0..15) = row; half hh holds K = hh*16 .. hh*16+15
// in elems 0..15.
__device__ __forceinline__ bf16x16 load_frag_a(const bf16* base, int ldk,
                                               int m0, int k0, int r, int hh) {
  const bf16* p = base + (size_t)(m0 + r) * (size_t)ldk + k0 + hh * 16;
  union { bf16x16 v; bf16x8 h[2]; } f;
  f.h[0] = *(const bf16x8*)(p);
  f.h[1] = *(const bf16x8*)(p + 8);
  return f.v;
}

// B fragment (32x16, KxN): lane r = column n; half hh holds K = hh*16 .. hh*16+15
// (contiguous along K in memory for NT layout).
__device__ __forceinline__ bf16x16 load_frag_b(const bf16* base, int ldk,
                                               int n0, int k0, int r, int hh) {
  const bf16* p = base + (size_t)(n0 + r) * (size_t)ldk + k0 + hh * 16;
  union { bf16x16 v; bf16x8 h[2]; } f;
  f.h[0] = *(const bf16x8*)(p);
  f.h[1] = *(const bf16x8*)(p + 8);
  return f.v;
}

// B fragment straight from an LDS tile where the 16 K-values per half are
// contiguous at p (+0 and +8 elements).
__device__ __forceinline__ bf16x16 frag_b_lin(const bf16* p) {
  union { bf16x16 v; bf16x8 h[2]; } f;
  f.h[0] = *(const bf16x8*)(p);
  f.h[1] = *(const bf16x8*)(p + 8);
  return f.v;
}

// 16-byte global -> LDS copy; async DMA on CDNA5 when available.
// the LDS address is taken from the VDST VGPR plus the offset; the low 32 bits of a generic
// LDS pointer are exactly the wave-relative LDS byte offset (aperture mapping).
__device__ __forceinline__ void cp16_g2l(const bf16* __restrict__ g, bf16* l) {
#if USE_ASYNC_LDS
  unsigned lds_off = (unsigned)(unsigned long long)(size_t)l;
  unsigned long long ga = (unsigned long long)(size_t)g;
  asm volatile("global_load_async_to_lds_b128 %0, %1, off"
               :
               : "v"(lds_off), "v"(ga)
               : "memory");
#else
  *(bf16x8*)l = *(const bf16x8*)g;
#endif
}

__device__ __forceinline__ void async_stage_wait() {
#if USE_ASYNC_LDS
  asm volatile("s_wait_asynccnt 0x0" ::: "memory");
#endif
}

// ---------------------------------------------------------------------------
// fp32 -> bf16 cast
// ---------------------------------------------------------------------------
__global__ void k_cast_bf16(const float* __restrict__ src, bf16* __restrict__ dst, int n) {
  int i = blockIdx.x * blockDim.x + threadIdx.x;
  if (i < n) dst[i] = (bf16)src[i];
}

// ---------------------------------------------------------------------------
// Generic NT GEMM:  C[M,N] = act( A[M,K] @ W[N,K]^T + bias )
//   op 0: store bf16 row-major
//   op 1: store bf16 row-major with exact GELU
//   op 2: store f32 row-major (outf)
//   op 3: store bf16 transposed per batch: out[b, n, m % batchRows]
// Block: 256 threads (8 waves). Block tile 128x128, wave tile 32x64.
// ---------------------------------------------------------------------------
__global__ __launch_bounds__(256)
void k_gemm_nt(const bf16* __restrict__ A, const bf16* __restrict__ W,
               const float* __restrict__ bias,
               bf16* __restrict__ outb, float* __restrict__ outf,
               int M, int N, int K, int op, int batchRows) {
  const int tid  = threadIdx.x;
  const int lane = tid & 31;
  const int wv   = tid >> 5;
  const int r    = lane & 15;
  const int hh   = lane >> 4;
  const int wrow = wv & 3;       // 4 waves along M
  const int wcol = wv >> 2;      // 2 waves along N
  const int m_base = blockIdx.y * 128 + wrow * 32;
  const int n_base = blockIdx.x * 128 + wcol * 64;

  f32x8 acc[2][4];
#pragma unroll
  for (int i = 0; i < 2; ++i)
#pragma unroll
    for (int j = 0; j < 4; ++j)
#pragma unroll
      for (int e = 0; e < 8; ++e) acc[i][j][e] = 0.0f;

  for (int k0 = 0; k0 < K; k0 += 32) {
    // prefetch next K-slab (speculative; dropped if OOB)
    __builtin_prefetch(A + (size_t)(m_base + r) * K + k0 + 64, 0, 1);
    __builtin_prefetch(W + (size_t)(n_base + r) * K + k0 + 64, 0, 1);

    bf16x16 a0 = load_frag_a(A, K, m_base,      k0, r, hh);
    bf16x16 a1 = load_frag_a(A, K, m_base + 16, k0, r, hh);
#pragma unroll
    for (int j = 0; j < 4; ++j) {
      bf16x16 bw = load_frag_b(W, K, n_base + 16 * j, k0, r, hh);
      acc[0][j] = wmma_bf16(a0, bw, acc[0][j]);
      acc[1][j] = wmma_bf16(a1, bw, acc[1][j]);
    }
  }

  // Epilogue. C layout: lanes 0-15 -> rows jj, lanes 16-31 -> rows jj+8; col = lane&15.
#pragma unroll
  for (int i = 0; i < 2; ++i)
#pragma unroll
    for (int j = 0; j < 4; ++j) {
      const int n  = n_base + 16 * j + r;
      const float bv = bias ? bias[n] : 0.0f;
#pragma unroll
      for (int jj = 0; jj < 8; ++jj) {
        const int m = m_base + 16 * i + jj + 8 * hh;
        float v = acc[i][j][jj] + bv;
        if (op == 1) v = 0.5f * v * (1.0f + erff(v * 0.70710678118654752f));
        if (op == 2) {
          outf[(size_t)m * N + n] = v;
        } else if (op == 3) {
          const int bidx = m / batchRows;
          const int tok  = m - bidx * batchRows;
          outb[((size_t)bidx * N + n) * (size_t)batchRows + tok] = (bf16)v;
        } else {
          outb[(size_t)m * N + n] = (bf16)v;
        }
      }
    }
}

// ---------------------------------------------------------------------------
// VAD head: out[row,0..2] = tanh( h[row,:] @ w2^T + b2 ), h is bf16 [rows,512]
// ---------------------------------------------------------------------------
__global__ void k_vad2(const bf16* __restrict__ h, const float* __restrict__ w2,
                       const float* __restrict__ b2, float* __restrict__ out, int rows) {
  int row = blockIdx.x * blockDim.x + threadIdx.x;
  if (row >= rows) return;
  const bf16* hp = h + (size_t)row * 512;
  float a0 = 0.f, a1 = 0.f, a2 = 0.f;
  for (int k = 0; k < 512; ++k) {
    float hv = (float)hp[k];
    a0 += hv * w2[k];
    a1 += hv * w2[512 + k];
    a2 += hv * w2[1024 + k];
  }
  out[(size_t)row * 3 + 0] = tanhf(a0 + b2[0]);
  out[(size_t)row * 3 + 1] = tanhf(a1 + b2[1]);
  out[(size_t)row * 3 + 2] = tanhf(a2 + b2[2]);
}

// ---------------------------------------------------------------------------
// Affinity bias: aff[b,q,k] = -lambda * || vq[b,q,:] - vk[b,k,:] ||_2
// ---------------------------------------------------------------------------
__global__ void k_aff(const float* __restrict__ vq, const float* __restrict__ vk,
                      const float* __restrict__ lam, float* __restrict__ aff,
                      int Bc, int Nq, int Nk) {
  size_t i = (size_t)blockIdx.x * blockDim.x + threadIdx.x;
  size_t tot = (size_t)Bc * Nq * Nk;
  if (i >= tot) return;
  int k = (int)(i % Nk);
  size_t t = i / Nk;
  int q = (int)(t % Nq);
  int b = (int)(t / Nq);
  const float* a = vq + ((size_t)b * Nq + q) * 3;
  const float* c = vk + ((size_t)b * Nk + k) * 3;
  float dx = a[0] - c[0], dy = a[1] - c[1], dz = a[2] - c[2];
  aff[i] = -lam[0] * sqrtf(dx * dx + dy * dy + dz * dz);
}

// ---------------------------------------------------------------------------
// Flash-style VAD-guided attention with cooperative LDS staging of K/V tiles
// (async global->LDS on CDNA5). grid: (NQ/64, B*H); block 128 = 4 waves.
// Qp, Kp: bf16 [B,N,D] row-major. VpT: bf16 [B, D, N]. ctx: bf16 [B,N,D].
// ---------------------------------------------------------------------------
#define HDIM 64
__global__ __launch_bounds__(128)
void k_attn(const bf16* __restrict__ Qp, const bf16* __restrict__ Kp,
            const bf16* __restrict__ VpT, const float* __restrict__ aff,
            bf16* __restrict__ ctx, int Hc, int Nc, float scale) {
  __shared__ __align__(16) bf16 klds[32 * 64];      // [key tok][hd]
  __shared__ __align__(16) bf16 vlds[64 * 32];      // [hd][key tok]
  __shared__ __align__(16) bf16 p_lds[4][16 * 32];  // per-wave P transpose

  const int tid  = threadIdx.x;
  const int lane = tid & 31;
  const int wv   = tid >> 5;
  const int r    = lane & 15;
  const int hh   = lane >> 4;
  const int bh = blockIdx.y;
  const int b  = bh / Hc;
  const int h  = bh % Hc;
  const int q0 = blockIdx.x * 64 + wv * 16;
  const int D  = Hc * HDIM;

  // Q fragments (kept in registers for the whole pass)
  const bf16* qbase = Qp + ((size_t)b * Nc + q0) * (size_t)D + h * HDIM;
  bf16x16 qa0 = load_frag_a(qbase, D, 0, 0,  r, hh);
  bf16x16 qa1 = load_frag_a(qbase, D, 0, 32, r, hh);

  f32x8 o[4];
#pragma unroll
  for (int t = 0; t < 4; ++t)
#pragma unroll
    for (int e = 0; e < 8; ++e) o[t][e] = 0.0f;

  float mrow[8], lrow[8];
#pragma unroll
  for (int jj = 0; jj < 8; ++jj) { mrow[jj] = -1e30f; lrow[jj] = 0.0f; }

  for (int nk0 = 0; nk0 < Nc; nk0 += 32) {
    // ---- cooperative staging of K (32x64) and V^T (64x32) tiles into LDS ----
    // 256 16B chunks each; 128 threads move 2 chunks per tile.
#pragma unroll
    for (int cc = 0; cc < 2; ++cc) {
      const int c   = tid + cc * 128;
      const int row = c >> 3;          // 8 chunks per 64-elem K row
      const int cin = c & 7;
      const bf16* g = Kp + ((size_t)b * Nc + nk0 + row) * (size_t)D + h * HDIM + cin * 8;
      cp16_g2l(g, &klds[row * 64 + cin * 8]);
    }
#pragma unroll
    for (int cc = 0; cc < 2; ++cc) {
      const int c   = tid + cc * 128;
      const int row = c >> 2;          // 4 chunks per 32-elem V^T row
      const int cin = c & 3;
      const bf16* g = VpT + ((size_t)b * D + h * HDIM + row) * (size_t)Nc + nk0 + cin * 8;
      cp16_g2l(g, &vlds[row * 32 + cin * 8]);
    }
    async_stage_wait();
    __syncthreads();

    // ---- scores S (16 x 32) = Q @ K^T for two 16-col tiles (K from LDS) ----
    f32x8 s[2];
#pragma unroll
    for (int t = 0; t < 2; ++t)
#pragma unroll
      for (int e = 0; e < 8; ++e) s[t][e] = 0.0f;

#pragma unroll
    for (int t = 0; t < 2; ++t) {
      const bf16* kp = &klds[(16 * t + r) * 64];
      bf16x16 kb0 = frag_b_lin(kp + 0  + hh * 16);
      bf16x16 kb1 = frag_b_lin(kp + 32 + hh * 16);
      s[t] = wmma_bf16(qa0, kb0, s[t]);
      s[t] = wmma_bf16(qa1, kb1, s[t]);
    }

    // ---- scale + VAD bias, per-row tile max ----
    float tmax[8];
#pragma unroll
    for (int jj = 0; jj < 8; ++jj) tmax[jj] = -1e30f;
#pragma unroll
    for (int t = 0; t < 2; ++t)
#pragma unroll
      for (int jj = 0; jj < 8; ++jj) {
        const int qrow = q0 + jj + 8 * hh;
        float v = s[t][jj] * scale +
                  aff[((size_t)b * Nc + qrow) * (size_t)Nc + nk0 + 16 * t + r];
        s[t][jj] = v;
        tmax[jj] = fmaxf(tmax[jj], v);
      }
#pragma unroll
    for (int mk = 1; mk < 16; mk <<= 1)
#pragma unroll
      for (int jj = 0; jj < 8; ++jj)
        tmax[jj] = fmaxf(tmax[jj], __shfl_xor(tmax[jj], mk, 32));

    float alpha[8];
#pragma unroll
    for (int jj = 0; jj < 8; ++jj) {
      float mn = fmaxf(mrow[jj], tmax[jj]);
      alpha[jj] = __expf(mrow[jj] - mn);
      mrow[jj] = mn;
    }

    // ---- P = exp(S - m), row sums ----
    float rsum[8];
#pragma unroll
    for (int jj = 0; jj < 8; ++jj) rsum[jj] = 0.0f;
#pragma unroll
    for (int t = 0; t < 2; ++t)
#pragma unroll
      for (int jj = 0; jj < 8; ++jj) {
        float p = __expf(s[t][jj] - mrow[jj]);
        s[t][jj] = p;
        rsum[jj] += p;
      }
#pragma unroll
    for (int mk = 1; mk < 16; mk <<= 1)
#pragma unroll
      for (int jj = 0; jj < 8; ++jj)
        rsum[jj] += __shfl_xor(rsum[jj], mk, 32);
#pragma unroll
    for (int jj = 0; jj < 8; ++jj) lrow[jj] = lrow[jj] * alpha[jj] + rsum[jj];

    // ---- cross-lane transpose C-layout -> A-layout via LDS (bf16) ----
#pragma unroll
    for (int t = 0; t < 2; ++t)
#pragma unroll
      for (int jj = 0; jj < 8; ++jj)
        p_lds[wv][(jj + 8 * hh) * 32 + 16 * t + r] = (bf16)s[t][jj];

    union { bf16x16 v; bf16x8 x[2]; } pf;
    pf.x[0] = *(const bf16x8*)&p_lds[wv][r * 32 + hh * 16];
    pf.x[1] = *(const bf16x8*)&p_lds[wv][r * 32 + hh * 16 + 8];

    // ---- rescale O, accumulate P @ V (V from LDS) ----
#pragma unroll
    for (int t4 = 0; t4 < 4; ++t4) {
#pragma unroll
      for (int jj = 0; jj < 8; ++jj) o[t4][jj] *= alpha[jj];
      bf16x16 vf = frag_b_lin(&vlds[(t4 * 16 + r) * 32 + hh * 16]);
      o[t4] = wmma_bf16(pf.v, vf, o[t4]);
    }

    __syncthreads();  // all waves done with klds/vlds before restaging
  }

  // ---- finalize: O /= l, store context ----
#pragma unroll
  for (int t4 = 0; t4 < 4; ++t4)
#pragma unroll
    for (int jj = 0; jj < 8; ++jj) {
      const int qrow = q0 + jj + 8 * hh;
      float v = o[t4][jj] / lrow[jj];
      ctx[((size_t)b * Nc + qrow) * (size_t)D + h * HDIM + t4 * 16 + r] = (bf16)v;
    }
}

// ---------------------------------------------------------------------------
// Fused residual + LayerNorm over D=1024, one block (256 threads) per row.
// ---------------------------------------------------------------------------
__global__ __launch_bounds__(256)
void k_ln(const float* __restrict__ proj, const float* __restrict__ resid,
          const float* __restrict__ g, const float* __restrict__ be,
          float* __restrict__ out, int Dm) {
  const int row = blockIdx.x;
  const int tid = threadIdx.x;
  __shared__ float red[256];

  float v[4];
  float s = 0.0f;
#pragma unroll
  for (int i = 0; i < 4; ++i) {
    const int c = tid + i * 256;
    float x = proj[(size_t)row * Dm + c] + resid[(size_t)row * Dm + c];
    v[i] = x;
    s += x;
  }
  red[tid] = s;
  __syncthreads();
  for (int o2 = 128; o2 > 0; o2 >>= 1) {
    if (tid < o2) red[tid] += red[tid + o2];
    __syncthreads();
  }
  const float mu = red[0] / (float)Dm;
  __syncthreads();

  float s2 = 0.0f;
#pragma unroll
  for (int i = 0; i < 4; ++i) { float d = v[i] - mu; s2 += d * d; }
  red[tid] = s2;
  __syncthreads();
  for (int o2 = 128; o2 > 0; o2 >>= 1) {
    if (tid < o2) red[tid] += red[tid + o2];
    __syncthreads();
  }
  const float var = red[0] / (float)Dm;
  const float rs = rsqrtf(var + 1e-5f);
#pragma unroll
  for (int i = 0; i < 4; ++i) {
    const int c = tid + i * 256;
    out[(size_t)row * Dm + c] = (v[i] - mu) * rs * g[c] + be[c];
  }
}

// ---------------------------------------------------------------------------
// Host launcher
// ---------------------------------------------------------------------------
extern "C" void kernel_launch(void* const* d_in, const int* in_sizes, int n_in,
                              void* d_out, int out_size, void* d_ws, size_t ws_size,
                              hipStream_t stream) {
  (void)in_sizes; (void)n_in; (void)out_size; (void)ws_size;

  constexpr int B = 4, NQ = 1024, NK = 1024, D = 1024, H = 16;
  constexpr int M = B * NQ;            // 4096 token rows
  constexpr float SCALE = 0.125f;      // (D/H)^-0.5 = 64^-0.5

  const float* query = (const float*)d_in[0];
  const float* key   = (const float*)d_in[1];
  const float* value = (const float*)d_in[2];
  const float* wq = (const float*)d_in[3];  const float* bq = (const float*)d_in[4];
  const float* wk = (const float*)d_in[5];  const float* bk = (const float*)d_in[6];
  const float* wvp = (const float*)d_in[7]; const float* bv = (const float*)d_in[8];
  const float* wo = (const float*)d_in[9];  const float* bo = (const float*)d_in[10];
  const float* w1 = (const float*)d_in[11]; const float* b1 = (const float*)d_in[12];
  const float* w2 = (const float*)d_in[13]; const float* b2 = (const float*)d_in[14];
  const float* lam  = (const float*)d_in[15];
  const float* ln_g = (const float*)d_in[16];
  const float* ln_b = (const float*)d_in[17];
  float* out = (float*)d_out;

  // workspace carve-up (256B aligned)
  size_t off = 0;
  char* base = (char*)d_ws;
  auto take = [&](size_t bytes) -> void* {
    void* p = base + off;
    off += (bytes + 255) & ~(size_t)255;
    return p;
  };
  bf16* xq  = (bf16*)take((size_t)M * D * 2);
  bf16* xk  = (bf16*)take((size_t)M * D * 2);
  bf16* xv  = (bf16*)take((size_t)M * D * 2);
  bf16* wqb = (bf16*)take((size_t)D * D * 2);
  bf16* wkb = (bf16*)take((size_t)D * D * 2);
  bf16* wvb = (bf16*)take((size_t)D * D * 2);
  bf16* wob = (bf16*)take((size_t)D * D * 2);
  bf16* w1b = (bf16*)take((size_t)(D / 2) * D * 2);
  bf16* Qp  = (bf16*)take((size_t)M * D * 2);
  bf16* Kp  = (bf16*)take((size_t)M * D * 2);
  bf16* VpT = (bf16*)take((size_t)M * D * 2);   // [B, D, N]
  bf16* hbuf = (bf16*)take((size_t)M * (D / 2) * 2);
  float* vq3 = (float*)take((size_t)M * 3 * 4);
  float* vk3 = (float*)take((size_t)M * 3 * 4);
  float* aff = (float*)take((size_t)B * NQ * NK * 4);
  bf16* ctx  = (bf16*)take((size_t)M * D * 2);
  float* proj = (float*)take((size_t)M * D * 4);

  // 1) casts to bf16
  const int nTok = M * D;        // 4,194,304
  const int nW   = D * D;        // 1,048,576
  const int nW1  = (D / 2) * D;  // 524,288
  k_cast_bf16<<<(nTok + 255) / 256, 256, 0, stream>>>(query, xq, nTok);
  k_cast_bf16<<<(nTok + 255) / 256, 256, 0, stream>>>(key,   xk, nTok);
  k_cast_bf16<<<(nTok + 255) / 256, 256, 0, stream>>>(value, xv, nTok);
  k_cast_bf16<<<(nW + 255) / 256, 256, 0, stream>>>(wq,  wqb, nW);
  k_cast_bf16<<<(nW + 255) / 256, 256, 0, stream>>>(wk,  wkb, nW);
  k_cast_bf16<<<(nW + 255) / 256, 256, 0, stream>>>(wvp, wvb, nW);
  k_cast_bf16<<<(nW + 255) / 256, 256, 0, stream>>>(wo,  wob, nW);
  k_cast_bf16<<<(nW1 + 255) / 256, 256, 0, stream>>>(w1, w1b, nW1);

  // 2) projections (WMMA GEMMs)
  dim3 gProj(D / 128, M / 128);   // (8, 32)
  k_gemm_nt<<<gProj, 256, 0, stream>>>(xq, wqb, bq, Qp,  nullptr, M, D, D, 0, NQ);
  k_gemm_nt<<<gProj, 256, 0, stream>>>(xk, wkb, bk, Kp,  nullptr, M, D, D, 0, NQ);
  k_gemm_nt<<<gProj, 256, 0, stream>>>(xv, wvb, bv, VpT, nullptr, M, D, D, 3, NQ);

  // 3) VAD MLP for query then key (hbuf reused; stream-ordered)
  dim3 gVad((D / 2) / 128, M / 128);  // (4, 32)
  k_gemm_nt<<<gVad, 256, 0, stream>>>(xq, w1b, b1, hbuf, nullptr, M, D / 2, D, 1, NQ);
  k_vad2<<<(M + 255) / 256, 256, 0, stream>>>(hbuf, w2, b2, vq3, M);
  k_gemm_nt<<<gVad, 256, 0, stream>>>(xk, w1b, b1, hbuf, nullptr, M, D / 2, D, 1, NQ);
  k_vad2<<<(M + 255) / 256, 256, 0, stream>>>(hbuf, w2, b2, vk3, M);

  // 4) VAD affinity bias (shared across heads)
  size_t totAff = (size_t)B * NQ * NK;
  k_aff<<<(unsigned)((totAff + 255) / 256), 256, 0, stream>>>(vq3, vk3, lam, aff, B, NQ, NK);

  // 5) flash attention with additive bias
  dim3 gAttn(NQ / 64, B * H);     // (16, 64)
  k_attn<<<gAttn, 128, 0, stream>>>(Qp, Kp, VpT, aff, ctx, H, NQ, SCALE);

  // 6) output projection (fp32 out) + residual LayerNorm
  k_gemm_nt<<<gProj, 256, 0, stream>>>(ctx, wob, bo, nullptr, proj, M, D, D, 2, NQ);
  k_ln<<<M, 256, 0, stream>>>(proj, query, ln_g, ln_b, out, D);
}
